// MatchAttentionLayer_1511828488809
// MI455X (gfx1250) — matrix operand order repaired and checked by reading, not compiled, not run on hardware
//
#include <hip/hip_runtime.h>

// ---------------- constants from the reference ----------------
#define Bb    2
#define Hh    48
#define Ww    96
#define Cc    128
#define NPix  4608          // H*W
#define MTOT  9216          // B*N
#define NHh   8
#define HD    32
#define ATTN_DIM 256
#define D_SELF 146
#define E_SELF 147
#define D_CROSS 130
#define HID 256
#define LDP 192             // padded row stride of projection outputs (sp/cp)
#define SCALE_ATT 0.17677669529663687f  // 32^-0.5

typedef _Float16 half_t;
typedef __attribute__((ext_vector_type(16))) _Float16 v16h;
typedef __attribute__((ext_vector_type(8)))  _Float16 v8h;
typedef __attribute__((ext_vector_type(8)))  float    v8f;

// ---------------- helpers ----------------
__device__ inline float warp_sum32(float v) {
  #pragma unroll
  for (int off = 16; off > 0; off >>= 1) v += __shfl_xor(v, off, 32);
  return v;
}

// ---------------- weight convert + transpose + pad: WT[n][k] (f16) ----------------
__global__ void k_wt_convert(half_t* __restrict__ dst, const float* __restrict__ src,
                             int Kin, int Kpad, int Nout, int NoutPad) {
  int i = blockIdx.x * blockDim.x + threadIdx.x;
  int tot = NoutPad * Kpad;
  if (i >= tot) return;
  int n = i / Kpad, k = i % Kpad;
  float v = (k < Kin && n < Nout) ? src[(size_t)k * Nout + n] : 0.0f;
  dst[i] = (half_t)v;
}

__global__ void k_f32_to_f16(half_t* __restrict__ dst, const float* __restrict__ src, int n) {
  int i = blockIdx.x * blockDim.x + threadIdx.x;
  if (i < n) dst[i] = (half_t)src[i];
}

// ---------------- WMMA GEMM: C[M x Nout] = A[M x K](f16) * W[K x Nout](f16, WT[n][k]) --------
// Each wave: 16x64 output tile (A-fragment reused across 4 WMMAs per K-step).
// act: 0=none, 1=silu. Optional bias[Nout], residual[M x ldres].
__global__ __launch_bounds__(128)
void k_gemm(const half_t* __restrict__ A, const half_t* __restrict__ WT,
            float* __restrict__ Cm, const float* __restrict__ bias,
            const float* __restrict__ residual, int ldres,
            int Nout, int K, int lda, int ldc, int act) {
  int wave = threadIdx.x >> 5;
  int lane = threadIdx.x & 31;
  int mtile = blockIdx.x * 4 + wave;
  int n0 = blockIdx.y * 64;
  int m0 = mtile * 16;
  int grp  = lane >> 4;      // 0 or 1
  int lidx = lane & 15;
  const half_t* arow = A  + (size_t)(m0 + lidx) * lda;
  const half_t* bc0  = WT + (size_t)(n0 +  0 + lidx) * K;
  const half_t* bc1  = WT + (size_t)(n0 + 16 + lidx) * K;
  const half_t* bc2  = WT + (size_t)(n0 + 32 + lidx) * K;
  const half_t* bc3  = WT + (size_t)(n0 + 48 + lidx) * K;

  v8f acc0 = {}; v8f acc1 = {}; v8f acc2 = {}; v8f acc3 = {};
  for (int k0 = 0; k0 < K; k0 += 32) {
    // A fragment (16x32 f16 layout): lane holds row m, K = grp*16 + e (contiguous)
    v8h alo = *(const v8h*)(arow + k0 + grp * 16);
    v8h ahi = *(const v8h*)(arow + k0 + grp * 16 + 8);
    v16h a;
    #pragma unroll
    for (int e = 0; e < 8; ++e) { a[e] = alo[e]; a[8 + e] = ahi[e]; }
    // B fragments (32x16): lane holds column n, K = grp*16 + e (contiguous in WT)
    v8h l0 = *(const v8h*)(bc0 + k0 + grp * 16);
    v8h h0 = *(const v8h*)(bc0 + k0 + grp * 16 + 8);
    v8h l1 = *(const v8h*)(bc1 + k0 + grp * 16);
    v8h h1 = *(const v8h*)(bc1 + k0 + grp * 16 + 8);
    v8h l2 = *(const v8h*)(bc2 + k0 + grp * 16);
    v8h h2 = *(const v8h*)(bc2 + k0 + grp * 16 + 8);
    v8h l3 = *(const v8h*)(bc3 + k0 + grp * 16);
    v8h h3 = *(const v8h*)(bc3 + k0 + grp * 16 + 8);
    v16h b0, b1, b2, b3;
    #pragma unroll
    for (int e = 0; e < 8; ++e) {
      b0[e] = l0[e]; b0[8 + e] = h0[e];
      b1[e] = l1[e]; b1[8 + e] = h1[e];
      b2[e] = l2[e]; b2[8 + e] = h2[e];
      b3[e] = l3[e]; b3[8 + e] = h3[e];
    }
    acc0 = __builtin_amdgcn_wmma_f32_16x16x32_f16(false, a, false, b0, (short)0, acc0, false, false);
    acc1 = __builtin_amdgcn_wmma_f32_16x16x32_f16(false, a, false, b1, (short)0, acc1, false, false);
    acc2 = __builtin_amdgcn_wmma_f32_16x16x32_f16(false, a, false, b2, (short)0, acc2, false, false);
    acc3 = __builtin_amdgcn_wmma_f32_16x16x32_f16(false, a, false, b3, (short)0, acc3, false, false);
  }
  // C/D layout: VGPR r -> M = m0 + grp*8 + r, N = n0 + t*16 + (lane&15)
  #pragma unroll
  for (int t = 0; t < 4; ++t) {
    v8f acc = (t == 0) ? acc0 : (t == 1) ? acc1 : (t == 2) ? acc2 : acc3;
    int nc = n0 + t * 16 + lidx;
    #pragma unroll
    for (int r = 0; r < 8; ++r) {
      int mr = m0 + grp * 8 + r;
      float v = acc[r];
      if (bias) v += bias[nc];
      if (act == 1) v = v / (1.0f + __expf(-v));
      if (residual) v += residual[(size_t)mr * ldres + nc];
      Cm[(size_t)mr * ldc + nc] = v;
    }
  }
}

// ---------------- consistency mask ----------------
__global__ void k_noc(const float* __restrict__ field, float* __restrict__ noc) {
  int m = blockIdx.x * blockDim.x + threadIdx.x;
  if (m >= MTOT) return;
  int b = m / NPix, n = m % NPix, y = n / Ww, xq = n % Ww;
  float fx_ = field[(size_t)m * 2 + 0];
  float fy_ = field[(size_t)m * 2 + 1];
  float xs = fminf(fmaxf(fx_ + (float)xq, 0.0f), (float)(Ww - 1));
  float ys = fminf(fmaxf(fy_ + (float)y,  0.0f), (float)(Hh - 1));
  float x0f = fminf(fmaxf(floorf(xs), 0.0f), (float)(Ww - 2));
  float y0f = fminf(fmaxf(floorf(ys), 0.0f), (float)(Hh - 2));
  int x0 = (int)x0f, y0 = (int)y0f;
  float wx = xs - x0f, wy = ys - y0f;
  const float* img = field + (size_t)(1 - b) * NPix * 2;
  const float* p00 = img + ((size_t)y0 * Ww + x0) * 2;
  const float* p01 = p00 + 2;
  const float* p10 = p00 + (size_t)Ww * 2;
  const float* p11 = p10 + 2;
  float vx = (1.f - wy) * ((1.f - wx) * p00[0] + wx * p01[0]) + wy * ((1.f - wx) * p10[0] + wx * p11[0]);
  float vy = (1.f - wy) * ((1.f - wx) * p00[1] + wx * p01[1]) + wy * ((1.f - wx) * p10[1] + wx * p11[1]);
  float diff = fabsf(fx_ + vx) + fabsf(fy_ + vy);
  noc[m] = (diff < 2.0f) ? 1.0f : 0.0f;
}

// ---------------- build xc + LN -> xe16 (padded 160) + self offsets ----------------
__global__ void k_build_xc_xe(const float* __restrict__ x, const float* __restrict__ rpos,
                              const float* __restrict__ field, const float* __restrict__ fs,
                              const float* __restrict__ n0g, const float* __restrict__ n0b,
                              const float* __restrict__ noc,
                              float* __restrict__ xc, half_t* __restrict__ xe16,
                              float* __restrict__ offs) {
  int warp = (blockIdx.x * blockDim.x + threadIdx.x) >> 5;
  int lane = threadIdx.x & 31;
  if (warp >= MTOT) return;
  int m = warp;
  int n = m % NPix, y = n / Ww, xq = n % Ww;
  float* xcr = xc + (size_t)m * D_SELF;
  float lv[5]; int cnt = 0;
  float s = 0.f, s2 = 0.f;
  for (int c = lane; c < D_SELF; c += 32) {
    float v;
    if (c < Cc)            v = x[(size_t)m * Cc + c];
    else if (c < Cc + 2)   v = field[(size_t)m * 2 + (c - Cc)] * fs[c - Cc];
    else                   v = rpos[(size_t)m * 16 + (c - (Cc + 2))];
    xcr[c] = v;
    lv[cnt++] = v; s += v; s2 += v * v;
  }
  if (lane < 16) {
    int comp = lane & 1;
    float base = comp ? (float)y : (float)xq;
    offs[(size_t)m * 16 + lane] = rpos[(size_t)m * 16 + lane] + base;
  }
  s = warp_sum32(s); s2 = warp_sum32(s2);
  float mu = s / (float)D_SELF;
  float var = s2 / (float)D_SELF - mu * mu;
  float rstd = rsqrtf(var + 1e-5f);
  half_t* xer = xe16 + (size_t)m * 160;
  cnt = 0;
  for (int c = lane; c < D_SELF; c += 32) {
    float hn = (lv[cnt++] - mu) * rstd * n0g[c] + n0b[c];
    xer[c] = (half_t)hn;
  }
  for (int c = D_SELF + lane; c < 160; c += 32)
    xer[c] = (c == D_SELF) ? (half_t)noc[m] : (half_t)0.0f;
}

// ---------------- match attention core (one thread per b,n,h) ----------------
__global__ void k_attention(const float* __restrict__ Q, const float* __restrict__ Kt,
                            const float* __restrict__ Vt, const float* __restrict__ offs,
                            float* __restrict__ O, float* __restrict__ ATTN) {
  int idx = blockIdx.x * blockDim.x + threadIdx.x;
  if (idx >= MTOT * NHh) return;
  int h = idx & 7;
  int m = idx >> 3;
  int b = m / NPix;
  float ox = offs[(size_t)m * 16 + h * 2 + 0];
  float oy = offs[(size_t)m * 16 + h * 2 + 1];
  ox = fminf(fmaxf(ox, 1.0f), (float)(Ww - 2) - 0.001f);
  oy = fminf(fmaxf(oy, 1.0f), (float)(Hh - 2) - 0.001f);
  float mxf = floorf(ox), myf = floorf(oy);
  float fx = ox - mxf, fy = oy - myf;
  int mx = (int)mxf, my = (int)myf;

  float q[HD];
  const float* qp = Q + (size_t)m * ATTN_DIM + h * HD;
  #pragma unroll
  for (int d = 0; d < HD; ++d) q[d] = qp[d];

  float S[16]; int kidx[16];
  for (int dy = 0; dy < 4; ++dy)
    for (int dx = 0; dx < 4; ++dx) {
      int a = dy * 4 + dx;
      int kn = (my - 1 + dy) * Ww + (mx - 1 + dx);
      kidx[a] = kn;
      const float* kp = Kt + ((size_t)(b * NPix + kn) * NHh + h) * HD;
      float sdot = 0.f;
      for (int d = 0; d < HD; ++d) sdot += q[d] * kp[d];
      S[a] = sdot * SCALE_ATT;
    }
  float w00 = (1.f - fy) * (1.f - fx), w01 = (1.f - fy) * fx;
  float w10 = fy * (1.f - fx),         w11 = fy * fx;
  float sc[9];
  #pragma unroll
  for (int i = 0; i < 3; ++i)
    #pragma unroll
    for (int j = 0; j < 3; ++j)
      sc[i * 3 + j] = w00 * S[i * 4 + j] + w01 * S[i * 4 + j + 1]
                    + w10 * S[(i + 1) * 4 + j] + w11 * S[(i + 1) * 4 + j + 1];
  float mmax = sc[0];
  #pragma unroll
  for (int i = 1; i < 9; ++i) mmax = fmaxf(mmax, sc[i]);
  float p[9], ssum = 0.f;
  #pragma unroll
  for (int i = 0; i < 9; ++i) { p[i] = __expf(sc[i] - mmax); ssum += p[i]; }
  float inv = 1.f / ssum;
  float Ag[16];
  #pragma unroll
  for (int a = 0; a < 16; ++a) Ag[a] = 0.f;
  #pragma unroll
  for (int i = 0; i < 3; ++i)
    #pragma unroll
    for (int j = 0; j < 3; ++j) {
      float pp = p[i * 3 + j] * inv;
      Ag[i * 4 + j]           += w00 * pp;
      Ag[i * 4 + j + 1]       += w01 * pp;
      Ag[(i + 1) * 4 + j]     += w10 * pp;
      Ag[(i + 1) * 4 + j + 1] += w11 * pp;
    }
  float outv[HD];
  #pragma unroll
  for (int d = 0; d < HD; ++d) outv[d] = 0.f;
  for (int a = 0; a < 16; ++a) {
    const float* vp = Vt + ((size_t)(b * NPix + kidx[a]) * NHh + h) * HD;
    float w = Ag[a];
    for (int d = 0; d < HD; ++d) outv[d] += w * vp[d];
  }
  float* op = O + (size_t)m * ATTN_DIM + h * HD;
  for (int d = 0; d < HD; ++d) op[d] = outv[d];
  if (ATTN) {
    float* ap = ATTN + ((size_t)m * NHh + h) * 16;
    #pragma unroll
    for (int a = 0; a < 16; ++a) ap[a] = Ag[a];
  }
}

// ---------------- post self-attention: residual, outputs, build cross inputs ----------------
__global__ void k_post_self(const float* __restrict__ xc, const float* __restrict__ P,
                            const float* __restrict__ fs,
                            const float* __restrict__ n1g, const float* __restrict__ n1b,
                            float* __restrict__ xc2, float* __restrict__ out1,
                            half_t* __restrict__ hA1, half_t* __restrict__ hA2,
                            float* __restrict__ offs) {
  int warp = (blockIdx.x * blockDim.x + threadIdx.x) >> 5;
  int lane = threadIdx.x & 31;
  if (warp >= MTOT) return;
  int m = warp;
  int b = m / NPix, n = m % NPix, y = n / Ww, xq = n % Ww;
  int mswap = (1 - b) * NPix + n;
  const float* xcr = xc + (size_t)m * D_SELF;
  const float* pr  = P  + (size_t)m * LDP;
  float* xc2r = xc2 + (size_t)m * D_CROSS;
  float lv[5]; int cnt = 0;
  float s = 0.f, s2 = 0.f;
  for (int c = lane; c < D_SELF; c += 32) {
    float v = xcr[c] + pr[c];
    if (c >= D_CROSS) {
      out1[(size_t)m * 16 + (c - D_CROSS)] = v;
    } else {
      if (c == D_CROSS - 1) v = 0.0f;
      xc2r[c] = v;
      lv[cnt++] = v; s += v; s2 += v * v;
    }
  }
  if (lane < 16) {
    int comp = lane & 1;
    float f1 = comp ? 0.0f : (xcr[Cc] + pr[Cc]) / fs[0];
    float base = comp ? (float)y : (float)xq;
    offs[(size_t)m * 16 + lane] = f1 + base;
  }
  s = warp_sum32(s); s2 = warp_sum32(s2);
  float mu = s / (float)D_CROSS;
  float var = s2 / (float)D_CROSS - mu * mu;
  float rstd = rsqrtf(var + 1e-5f);
  cnt = 0;
  for (int c = lane; c < D_CROSS; c += 32) {
    float hn = (lv[cnt++] - mu) * rstd * n1g[c] + n1b[c];
    hA1[(size_t)m * 160 + c]     = (half_t)hn;
    hA2[(size_t)mswap * 160 + c] = (half_t)hn;   // tgt = swap_halves(ref)
  }
  for (int c = D_CROSS + lane; c < 160; c += 32) {
    hA1[(size_t)m * 160 + c]     = (half_t)0.0f;
    hA2[(size_t)mswap * 160 + c] = (half_t)0.0f;
  }
}

// ---------------- build ocat16 = [g*o (256) | attn (128)] ----------------
__global__ void k_build_ocat(const float* __restrict__ G, const float* __restrict__ O,
                             const float* __restrict__ ATTN, half_t* __restrict__ hB) {
  int i = blockIdx.x * blockDim.x + threadIdx.x;
  if (i >= MTOT * 384) return;
  int m = i / 384, c = i % 384;
  float v;
  if (c < 256) v = G[(size_t)m * 256 + c] * O[(size_t)m * 256 + c];
  else         v = ATTN[(size_t)m * 128 + (c - 256)];
  hB[i] = (half_t)v;
}

// ---------------- post cross-attention: residual, field2 out, LN -> h2 ----------------
__global__ void k_post_cross(float* __restrict__ xc2, const float* __restrict__ P,
                             const float* __restrict__ fs,
                             const float* __restrict__ n2g, const float* __restrict__ n2b,
                             float* __restrict__ out2, half_t* __restrict__ h2) {
  int warp = (blockIdx.x * blockDim.x + threadIdx.x) >> 5;
  int lane = threadIdx.x & 31;
  if (warp >= MTOT) return;
  int m = warp;
  float* xc2r = xc2 + (size_t)m * D_CROSS;
  const float* pr = P + (size_t)m * LDP;
  float lv[4]; int cnt = 0;
  float s = 0.f, s2 = 0.f;
  float f2x = 0.f;
  for (int c = lane; c < D_CROSS; c += 32) {
    float v = xc2r[c] + pr[c];
    if (c == D_CROSS - 1) v = 0.0f;
    xc2r[c] = v;
    if (c == Cc) f2x = v;
    if (c < Cc) { lv[cnt++] = v; s += v; s2 += v * v; }
  }
  if (lane == 0) {
    out2[(size_t)m * 2 + 0] = f2x / fs[0];
    out2[(size_t)m * 2 + 1] = 0.0f;
  }
  s = warp_sum32(s); s2 = warp_sum32(s2);
  float mu = s / (float)Cc;
  float var = s2 / (float)Cc - mu * mu;
  float rstd = rsqrtf(var + 1e-5f);
  cnt = 0;
  for (int c = lane; c < Cc; c += 32) {
    float hn = (lv[cnt++] - mu) * rstd * n2g[c] + n2b[c];
    h2[(size_t)m * Cc + c] = (half_t)hn;
  }
}

// ---------------- depthwise 3x3 conv + gelu * gate -> f16 ----------------
__global__ void k_dwconv(const float* __restrict__ U, const float* __restrict__ dww,
                         const float* __restrict__ dwb, half_t* __restrict__ act16) {
  int i = blockIdx.x * blockDim.x + threadIdx.x;
  if (i >= MTOT * HID) return;
  int c = i & 255;
  int m = i >> 8;
  int b = m / NPix, n = m % NPix, y = n / Ww, xq = n % Ww;
  float acc = dwb[c];
  #pragma unroll
  for (int ky = 0; ky < 3; ++ky) {
    int yy = y + ky - 1;
    if (yy < 0 || yy >= Hh) continue;
    #pragma unroll
    for (int kx = 0; kx < 3; ++kx) {
      int xx = xq + kx - 1;
      if (xx < 0 || xx >= Ww) continue;
      size_t um = (size_t)(b * NPix + yy * Ww + xx);
      acc += U[um * 512 + c] * dww[(ky * 3 + kx) * HID + c];
    }
  }
  float g = 0.5f * acc * (1.0f + erff(acc * 0.70710678118654752f));
  float gv = U[(size_t)m * 512 + 256 + c];
  act16[(size_t)m * HID + c] = (half_t)(g * gv);
}

// ---------------- host launcher ----------------
extern "C" void kernel_launch(void* const* d_in, const int* in_sizes, int n_in,
                              void* d_out, int out_size, void* d_ws, size_t ws_size,
                              hipStream_t stream) {
  const float* x      = (const float*)d_in[0];
  const float* rpos   = (const float*)d_in[1];
  const float* field  = (const float*)d_in[2];
  const float* fs     = (const float*)d_in[3];
  const float* n0g = (const float*)d_in[4];  const float* n0b = (const float*)d_in[5];
  const float* n1g = (const float*)d_in[6];  const float* n1b = (const float*)d_in[7];
  const float* n2g = (const float*)d_in[8];  const float* n2b = (const float*)d_in[9];
  const float* sq_w = (const float*)d_in[10];
  const float* sk_w = (const float*)d_in[11];
  const float* sv_w = (const float*)d_in[12];
  const float* sp_w = (const float*)d_in[13];
  const float* cq_w = (const float*)d_in[14];
  const float* ck_w = (const float*)d_in[15];
  const float* cv_w = (const float*)d_in[16];
  const float* cg_w = (const float*)d_in[17];
  const float* cp_w = (const float*)d_in[18];
  const float* fc1_w = (const float*)d_in[19];
  const float* fc1_b = (const float*)d_in[20];
  const float* dw_w  = (const float*)d_in[21];
  const float* dw_b  = (const float*)d_in[22];
  const float* fc2_w = (const float*)d_in[23];
  const float* fc2_b = (const float*)d_in[24];

  float* out0 = (float*)d_out;                       // xo         (M*128)
  float* out1 = (float*)d_out + (size_t)MTOT * 128;  // self_rpos  (M*16)
  float* out2 = (float*)d_out + (size_t)MTOT * 144;  // field2     (M*2)

  // -------- workspace bump allocator (256B aligned) --------
  char* ws = (char*)d_ws;
  size_t off = 0;
  auto alloc = [&](size_t bytes) -> void* {
    void* p = ws + off;
    off = (off + bytes + 255) & ~(size_t)255;
    return p;
  };
  float*  f_noc  = (float*)alloc((size_t)MTOT * 4);
  float*  f_xc   = (float*)alloc((size_t)MTOT * D_SELF * 4);
  float*  f_Q    = (float*)alloc((size_t)MTOT * 256 * 4);
  float*  f_K    = (float*)alloc((size_t)MTOT * 256 * 4);
  float*  f_V    = (float*)alloc((size_t)MTOT * 256 * 4);
  float*  f_O    = (float*)alloc((size_t)MTOT * 256 * 4);
  float*  f_P    = (float*)alloc((size_t)MTOT * LDP * 4);
  float*  f_G    = (float*)alloc((size_t)MTOT * 256 * 4);
  float*  f_ATT  = (float*)alloc((size_t)MTOT * 128 * 4);
  float*  f_XC2  = (float*)alloc((size_t)MTOT * D_CROSS * 4);
  float*  f_U    = (float*)alloc((size_t)MTOT * 512 * 4);
  float*  f_OFFS = (float*)alloc((size_t)MTOT * 16 * 4);
  half_t* hA1    = (half_t*)alloc((size_t)MTOT * 160 * 2);
  half_t* hA2    = (half_t*)alloc((size_t)MTOT * 160 * 2);
  half_t* hB     = (half_t*)alloc((size_t)MTOT * 384 * 2);
  half_t* hC     = (half_t*)alloc((size_t)MTOT * 256 * 2);
  half_t* sqT  = (half_t*)alloc((size_t)256 * 160 * 2);
  half_t* skT  = (half_t*)alloc((size_t)256 * 160 * 2);
  half_t* svT  = (half_t*)alloc((size_t)256 * 160 * 2);
  half_t* spT  = (half_t*)alloc((size_t)LDP * 256 * 2);
  half_t* cqT  = (half_t*)alloc((size_t)256 * 160 * 2);
  half_t* ckT  = (half_t*)alloc((size_t)256 * 160 * 2);
  half_t* cvT  = (half_t*)alloc((size_t)256 * 160 * 2);
  half_t* cgT  = (half_t*)alloc((size_t)256 * 160 * 2);
  half_t* cpT  = (half_t*)alloc((size_t)LDP * 384 * 2);
  half_t* fc1T = (half_t*)alloc((size_t)512 * 128 * 2);
  half_t* fc2T = (half_t*)alloc((size_t)128 * 256 * 2);

  auto wconv = [&](half_t* dst, const float* src, int Kin, int Kpad, int Nout, int NoutPad) {
    int tot = NoutPad * Kpad;
    k_wt_convert<<<dim3((tot + 255) / 256), dim3(256), 0, stream>>>(dst, src, Kin, Kpad, Nout, NoutPad);
  };
  auto gemm = [&](const half_t* A, int lda, const half_t* WT, float* Cm, int ldc,
                  int Nout, int Kp, const float* bias, int act,
                  const float* res, int ldres) {
    dim3 g(MTOT / 64, Nout / 64), blk(128);
    k_gemm<<<g, blk, 0, stream>>>(A, WT, Cm, bias, res, ldres, Nout, Kp, lda, ldc, act);
  };

  // -------- weights -> f16 transposed/padded --------
  wconv(sqT,  sq_w,  E_SELF, 160, 256, 256);
  wconv(skT,  sk_w,  E_SELF, 160, 256, 256);
  wconv(svT,  sv_w,  E_SELF, 160, 256, 256);
  wconv(spT,  sp_w,  256,    256, D_SELF, LDP);
  wconv(cqT,  cq_w,  D_CROSS, 160, 256, 256);
  wconv(ckT,  ck_w,  D_CROSS, 160, 256, 256);
  wconv(cvT,  cv_w,  D_CROSS, 160, 256, 256);
  wconv(cgT,  cg_w,  D_CROSS, 160, 256, 256);
  wconv(cpT,  cp_w,  384,    384, D_CROSS, LDP);
  wconv(fc1T, fc1_w, 128,    128, 512, 512);
  wconv(fc2T, fc2_w, 256,    256, 128, 128);

  // -------- stage 1: consistency mask, xc/LN/xe16, self offsets --------
  k_noc<<<dim3((MTOT + 255) / 256), dim3(256), 0, stream>>>(field, f_noc);
  k_build_xc_xe<<<dim3(MTOT / 8), dim3(256), 0, stream>>>(x, rpos, field, fs, n0g, n0b,
                                                          f_noc, f_xc, hA1, f_OFFS);
  // -------- self QKV + attention + output proj --------
  gemm(hA1, 160, sqT, f_Q, 256, 256, 160, nullptr, 0, nullptr, 0);
  gemm(hA1, 160, skT, f_K, 256, 256, 160, nullptr, 0, nullptr, 0);
  gemm(hA1, 160, svT, f_V, 256, 256, 160, nullptr, 0, nullptr, 0);
  k_attention<<<dim3(MTOT * NHh / 256), dim3(256), 0, stream>>>(f_Q, f_K, f_V, f_OFFS, f_O, (float*)nullptr);
  k_f32_to_f16<<<dim3(MTOT), dim3(256), 0, stream>>>(hC, f_O, MTOT * 256);
  gemm(hC, 256, spT, f_P, LDP, LDP, 256, nullptr, 0, nullptr, 0);
  k_post_self<<<dim3(MTOT / 8), dim3(256), 0, stream>>>(f_xc, f_P, fs, n1g, n1b,
                                                        f_XC2, out1, hA1, hA2, f_OFFS);
  // -------- cross QKVG + attention + output proj --------
  gemm(hA1, 160, cqT, f_Q, 256, 256, 160, nullptr, 0, nullptr, 0);
  gemm(hA2, 160, ckT, f_K, 256, 256, 160, nullptr, 0, nullptr, 0);
  gemm(hA2, 160, cvT, f_V, 256, 256, 160, nullptr, 0, nullptr, 0);
  gemm(hA1, 160, cgT, f_G, 256, 256, 160, nullptr, 1 /*silu*/, nullptr, 0);
  k_attention<<<dim3(MTOT * NHh / 256), dim3(256), 0, stream>>>(f_Q, f_K, f_V, f_OFFS, f_O, f_ATT);
  k_build_ocat<<<dim3((MTOT * 384 + 255) / 256), dim3(256), 0, stream>>>(f_G, f_O, f_ATT, hB);
  gemm(hB, 384, cpT, f_P, LDP, LDP, 384, nullptr, 0, nullptr, 0);
  k_post_cross<<<dim3(MTOT / 8), dim3(256), 0, stream>>>(f_XC2, f_P, fs, n2g, n2b, out2, hB);
  // -------- MLP: fc1 (+bias) -> dwconv/gelu/gate -> fc2 (+bias +residual) --------
  gemm(hB, 128, fc1T, f_U, 512, 512, 128, fc1_b, 0, nullptr, 0);
  k_dwconv<<<dim3(MTOT), dim3(256), 0, stream>>>(f_U, dw_w, dw_b, hC);
  gemm(hC, 256, fc2T, out0, 128, 128, 256, fc2_b, 0, f_XC2, D_CROSS);

  (void)in_sizes; (void)n_in; (void)out_size; (void)ws_size;
}
